// MultiHeadAttention_36318243455145
// MI455X (gfx1250) — matrix operand order repaired and checked by reading, not compiled, not run on hardware
//
#include <hip/hip_runtime.h>

// ---------------------------------------------------------------------------
// MHA forward for MI455X (gfx1250, wave32, WMMA bf16 16x16x32, f32 accum)
// Attention K/V staging uses async global->LDS copies (ASYNCcnt) with
// double-buffered LDS tiles.
// ---------------------------------------------------------------------------

typedef __attribute__((ext_vector_type(16))) __bf16        v16bf;
typedef __attribute__((ext_vector_type(8)))  float         v8f;
typedef __attribute__((ext_vector_type(8)))  unsigned int  v8u;

#define D_MODEL 1024
#define SEQ     2048
#define BATCH   2
#define NHEAD   16
#define DH      64
#define ROWS    (BATCH*SEQ)   // 4096 tokens

// ----- f32 -> bf16 round-to-nearest-even ------------------------------------
__device__ __forceinline__ unsigned short f2bf(float f) {
  unsigned int u = __float_as_uint(f);
  u += 0x7FFFu + ((u >> 16) & 1u);
  return (unsigned short)(u >> 16);
}
__device__ __forceinline__ unsigned int pack2bf(float lo, float hi) {
  return (unsigned int)f2bf(lo) | ((unsigned int)f2bf(hi) << 16);
}

// ----- WMMA fragment loaders (contiguous k = 16h + s order, same for A and B) -----
__device__ __forceinline__ v16bf frag_2x16B(const unsigned short* p0,
                                            const unsigned short* p1) {
  uint4 x = *(const uint4*)p0;
  uint4 y = *(const uint4*)p1;
  v8u r;
  r[0]=x.x; r[1]=x.y; r[2]=x.z; r[3]=x.w;
  r[4]=y.x; r[5]=y.y; r[6]=y.z; r[7]=y.w;
  return __builtin_bit_cast(v16bf, r);
}
__device__ __forceinline__ v16bf frag_32B(const unsigned short* p) {
  return frag_2x16B(p, p + 8);
}

__device__ __forceinline__ v8f wmma_bf16(v16bf a, v16bf b, v8f c) {
  return __builtin_amdgcn_wmma_f32_16x16x32_bf16(false, a, false, b,
                                                 (short)0, c, false, false);
}

__device__ __forceinline__ v8f zero8() {
  v8f z = {0.f,0.f,0.f,0.f,0.f,0.f,0.f,0.f};
  return z;
}

// ----- async global -> LDS (tracked by ASYNCcnt, bypasses VGPRs) ------------
__device__ __forceinline__ void async_ld_b128(unsigned short* lds_dst,
                                              const unsigned short* gsrc) {
  unsigned int laddr = (unsigned int)(size_t)lds_dst;  // flat low 32b = LDS addr
  asm volatile("global_load_async_to_lds_b128 %0, %1, off"
               :: "v"(laddr), "v"(gsrc) : "memory");
}
template <int N>
__device__ __forceinline__ void wait_asynccnt() {
  asm volatile("s_wait_asynccnt %0" :: "i"(N) : "memory");
}

// ===========================================================================
// Kernel 1: fused Q/K/V projection  C = X @ W^T + b   (bf16 out, f32 acc)
//   grid (ROWS/128, D/64, 3), block 256 (8 waves); z: 0=Q(scaled) 1=K 2=V(T)
// ===========================================================================
__global__ __launch_bounds__(256) void qkv_proj_kernel(
    const float* __restrict__ q,  const float* __restrict__ k,
    const float* __restrict__ v,
    const float* __restrict__ Wq, const float* __restrict__ bq,
    const float* __restrict__ Wk, const float* __restrict__ bk,
    const float* __restrict__ Wv, const float* __restrict__ bv,
    unsigned short* __restrict__ Qh, unsigned short* __restrict__ Kh,
    unsigned short* __restrict__ Vt)
{
  __shared__ __align__(16) unsigned short aT[128 * 32];  // [tokRow][k] bf16
  __shared__ __align__(16) unsigned short bT[64 * 32];   // [featRow][k] bf16

  const int z = blockIdx.z;
  const float* X    = (z == 0) ? q  : (z == 1) ? k  : v;
  const float* W    = (z == 0) ? Wq : (z == 1) ? Wk : Wv;
  const float* bias = (z == 0) ? bq : (z == 1) ? bk : bv;

  const int mbase = blockIdx.x * 128;
  const int nbase = blockIdx.y * 64;
  const int tid  = threadIdx.x;
  const int wave = tid >> 5;
  const int lane = tid & 31;
  const int l16  = lane & 15;
  const bool hif = lane >= 16;

  v8f acc[4];
  for (int i = 0; i < 4; i++) acc[i] = zero8();

  for (int kb = 0; kb < D_MODEL; kb += 32) {
    // stage A tile 128x32 (f32 -> bf16): 1024 float4, 4 per thread
    for (int i = 0; i < 4; i++) {
      int idx = tid + i * 256;
      int row = idx >> 3, c4 = idx & 7;
      float4 f = *(const float4*)(X + (size_t)(mbase + row) * D_MODEL + kb + c4 * 4);
      *(uint2*)(aT + row * 32 + c4 * 4) =
          make_uint2(pack2bf(f.x, f.y), pack2bf(f.z, f.w));
    }
    // stage B tile 64x32 (weight rows, contiguous along k): 512 float4
    for (int i = 0; i < 2; i++) {
      int idx = tid + i * 256;
      int row = idx >> 3, c4 = idx & 7;
      float4 f = *(const float4*)(W + (size_t)(nbase + row) * D_MODEL + kb + c4 * 4);
      *(uint2*)(bT + row * 32 + c4 * 4) =
          make_uint2(pack2bf(f.x, f.y), pack2bf(f.z, f.w));
    }
    __syncthreads();

    const unsigned short* ap = aT + (wave * 16 + l16) * 32 + (hif ? 16 : 0);
    v16bf a = frag_32B(ap);
    for (int nt = 0; nt < 4; nt++) {
      v16bf b = frag_32B(bT + (nt * 16 + l16) * 32 + (hif ? 16 : 0));
      acc[nt] = wmma_bf16(a, b, acc[nt]);
    }
    __syncthreads();
  }

  // epilogue: bias (+ 1/sqrt(dh) folded into Q), head-split layouts
  const int mrow = mbase + wave * 16 + (hif ? 8 : 0);
  for (int nt = 0; nt < 4; nt++) {
    int n = nbase + nt * 16 + l16;
    int h = n >> 6, d = n & 63;
    float bi = bias[n];
    if (z < 2) {
      unsigned short* out = (z == 0) ? Qh : Kh;
      float sc = (z == 0) ? 0.125f : 1.0f;  // 1/sqrt(64)
      for (int vv = 0; vv < 8; vv++) {
        int m = mrow + vv;
        int b2 = m >> 11, ss = m & 2047;
        out[(((size_t)(b2 * NHEAD + h)) * SEQ + ss) * DH + d] =
            f2bf((acc[nt][vv] + bi) * sc);
      }
    } else {
      // V stored transposed [B,H,dh,S]; 8 consecutive tokens -> one b128 store
      int b2 = mrow >> 11, ss = mrow & 2047;
      unsigned int w0 = pack2bf(acc[nt][0] + bi, acc[nt][1] + bi);
      unsigned int w1 = pack2bf(acc[nt][2] + bi, acc[nt][3] + bi);
      unsigned int w2 = pack2bf(acc[nt][4] + bi, acc[nt][5] + bi);
      unsigned int w3 = pack2bf(acc[nt][6] + bi, acc[nt][7] + bi);
      *(uint4*)(Vt + (((size_t)(b2 * NHEAD + h)) * DH + d) * SEQ + ss) =
          make_uint4(w0, w1, w2, w3);
    }
  }
}

// ===========================================================================
// Kernel 2: flash attention, transposed-score formulation, async staging.
//   S^T = K * Q^T  (contraction over dh) -> softmax over keys is per-lane
//   O^T += V^T * P^T (contraction over keys), online max/sum rescaling.
//   K/V tiles double-buffered in LDS via global_load_async_to_lds_b128.
//   grid (SEQ/128, NHEAD, BATCH), block 256 (8 waves; 16 q-rows per wave)
// ===========================================================================
__global__ __launch_bounds__(256) void attn_kernel(
    const unsigned short* __restrict__ Qh, const unsigned short* __restrict__ Kh,
    const unsigned short* __restrict__ Vt, unsigned short* __restrict__ Abuf)
{
  __shared__ __align__(16) unsigned short kbuf[2][32 * 64];  // [buf][key][dh]
  __shared__ __align__(16) unsigned short vbuf[2][64 * 32];  // [buf][dh][key]

  const int b = blockIdx.z, h = blockIdx.y;
  const int bh = b * NHEAD + h;
  const int tid = threadIdx.x, wave = tid >> 5, lane = tid & 31, l16 = lane & 15;
  const bool hif = lane >= 16;
  const int qbase = blockIdx.x * 128 + wave * 16;

  const unsigned short* Kbh = Kh + (size_t)bh * SEQ * DH;
  const unsigned short* Vbh = Vt + (size_t)bh * DH * SEQ;

  // per-thread staging coordinates (each thread: one b128 of K, one of V)
  const int kkk = tid >> 3, kd8 = tid & 7;   // K: key row, dh octet
  const int vdd = tid >> 2, vk8 = tid & 3;   // V: dh row, key octet

  // issue async copy of one 32-key block into buffer `bi` (2 ops/thread/wave)
  auto issue_block = [&](int kb2, int bi) {
    async_ld_b128(&kbuf[bi][kkk * 64 + kd8 * 8],
                  Kbh + (size_t)(kb2 + kkk) * DH + kd8 * 8);
    async_ld_b128(&vbuf[bi][vdd * 32 + vk8 * 8],
                  Vbh + (size_t)vdd * SEQ + kb2 + vk8 * 8);
  };

  // Q as B-operand: lane = q column, contiguous dh; loaded once, reused 64x.
  const unsigned short* qrow = Qh + ((size_t)bh * SEQ + qbase + l16) * DH;
  v16bf bq0 = frag_32B(qrow + (hif ? 16 : 0));        // dh 0..31
  v16bf bq1 = frag_32B(qrow + 32 + (hif ? 16 : 0));   // dh 32..63

  v8f oacc[4];
  for (int i = 0; i < 4; i++) oacc[i] = zero8();
  float m_run = -1e30f, l_run = 0.0f;

  constexpr int NBLK = SEQ / 32;  // 64
  issue_block(0, 0);
  issue_block(32, 1);

  for (int it = 0; it < NBLK; it++) {
    const int cur = it & 1;
    // wait for this block's copies; leave next block's pair in flight
    if (it < NBLK - 1) wait_asynccnt<2>(); else wait_asynccnt<0>();
    __syncthreads();

    const unsigned short* kb_l = kbuf[cur];
    const unsigned short* vb_l = vbuf[cur];

    // scores: X0 = keys 0..15, X1 = keys 16..31 (C-layout: lane=q, vgpr=key)
    v8f x0 = zero8(), x1 = zero8();
    {
      const unsigned short* base = kb_l + l16 * 64 + (hif ? 16 : 0);
      x0 = wmma_bf16(frag_2x16B(base,      base + 8), bq0, x0);
      x0 = wmma_bf16(frag_2x16B(base + 32, base + 40), bq1, x0);
      const unsigned short* base1 = base + 16 * 64;
      x1 = wmma_bf16(frag_2x16B(base1,      base1 + 8), bq0, x1);
      x1 = wmma_bf16(frag_2x16B(base1 + 32, base1 + 40), bq1, x1);
    }

    // online softmax over the 32 keys of this block (per q column)
    float mx = x0[0];
    for (int i = 1; i < 8; i++) mx = fmaxf(mx, x0[i]);
    for (int i = 0; i < 8; i++) mx = fmaxf(mx, x1[i]);
    mx = fmaxf(mx, __shfl_xor(mx, 16));
    float m_new = fmaxf(m_run, mx);
    float scale = __expf(m_run - m_new);
    float p0[8], p1[8], s = 0.f;
    for (int i = 0; i < 8; i++) { p0[i] = __expf(x0[i] - m_new); s += p0[i]; }
    for (int i = 0; i < 8; i++) { p1[i] = __expf(x1[i] - m_new); s += p1[i]; }
    s += __shfl_xor(s, 16);
    l_run = l_run * scale + s;
    m_run = m_new;
    for (int i = 0; i < 4; i++)
      for (int j = 0; j < 8; j++) oacc[i][j] *= scale;

    // re-lay P (C-layout) into B-fragment layout via half-wave swaps
    unsigned int P0[4], P1[4];
    for (int j = 0; j < 4; j++) {
      P0[j] = pack2bf(p0[2 * j], p0[2 * j + 1]);
      P1[j] = pack2bf(p1[2 * j], p1[2 * j + 1]);
    }
    v8u bp;
    for (int j = 0; j < 4; j++) {
      unsigned int t0 = __shfl_xor(P0[j], 16);
      unsigned int t1 = __shfl_xor(P1[j], 16);
      bp[j]     = hif ? t1 : P0[j];
      bp[j + 4] = hif ? P1[j] : t0;
    }
    v16bf bP = __builtin_bit_cast(v16bf, bp);

    // O^T += V^T * P^T : 4 dh-tiles, contraction over 32 keys
    for (int mt = 0; mt < 4; mt++) {
      const unsigned short* vb = vb_l + (mt * 16 + l16) * 32 + (hif ? 16 : 0);
      oacc[mt] = wmma_bf16(frag_2x16B(vb, vb + 8), bP, oacc[mt]);
    }
    __syncthreads();  // all waves done reading buf[cur] before refill

    if (it + 2 < NBLK) issue_block((it + 2) * 32, cur);
  }

  // finalize: divide by softmax sum, write [B,S,D] bf16 (8 consec d per store)
  float inv = 1.0f / l_run;
  int qg = qbase + l16;
  unsigned short* orow = Abuf + ((size_t)b * SEQ + qg) * D_MODEL + h * DH;
  for (int mt = 0; mt < 4; mt++) {
    unsigned int w0 = pack2bf(oacc[mt][0] * inv, oacc[mt][1] * inv);
    unsigned int w1 = pack2bf(oacc[mt][2] * inv, oacc[mt][3] * inv);
    unsigned int w2 = pack2bf(oacc[mt][4] * inv, oacc[mt][5] * inv);
    unsigned int w3 = pack2bf(oacc[mt][6] * inv, oacc[mt][7] * inv);
    *(uint4*)(orow + mt * 16 + (hif ? 8 : 0)) = make_uint4(w0, w1, w2, w3);
  }
}

// ===========================================================================
// Kernel 3: output projection  out = A @ Wo^T + bo   (f32 out)
//   grid (ROWS/128, D/64), block 256
// ===========================================================================
__global__ __launch_bounds__(256) void out_proj_kernel(
    const unsigned short* __restrict__ Abuf, const float* __restrict__ Wo,
    const float* __restrict__ bo, float* __restrict__ out)
{
  __shared__ __align__(16) unsigned short aT[128 * 32];
  __shared__ __align__(16) unsigned short bT[64 * 32];

  const int mbase = blockIdx.x * 128;
  const int nbase = blockIdx.y * 64;
  const int tid  = threadIdx.x;
  const int wave = tid >> 5;
  const int lane = tid & 31;
  const int l16  = lane & 15;
  const bool hif = lane >= 16;

  v8f acc[4];
  for (int i = 0; i < 4; i++) acc[i] = zero8();

  for (int kb = 0; kb < D_MODEL; kb += 32) {
    // A tile already bf16: 512 uint4, 2 per thread
    for (int i = 0; i < 2; i++) {
      int idx = tid + i * 256;
      int row = idx >> 2, c8 = idx & 3;
      *(uint4*)(aT + row * 32 + c8 * 8) =
          *(const uint4*)(Abuf + (size_t)(mbase + row) * D_MODEL + kb + c8 * 8);
    }
    // B tile: Wo rows f32 -> bf16
    for (int i = 0; i < 2; i++) {
      int idx = tid + i * 256;
      int row = idx >> 3, c4 = idx & 7;
      float4 f = *(const float4*)(Wo + (size_t)(nbase + row) * D_MODEL + kb + c4 * 4);
      *(uint2*)(bT + row * 32 + c4 * 4) =
          make_uint2(pack2bf(f.x, f.y), pack2bf(f.z, f.w));
    }
    __syncthreads();

    const unsigned short* ap = aT + (wave * 16 + l16) * 32 + (hif ? 16 : 0);
    v16bf a = frag_32B(ap);
    for (int nt = 0; nt < 4; nt++) {
      v16bf b = frag_32B(bT + (nt * 16 + l16) * 32 + (hif ? 16 : 0));
      acc[nt] = wmma_bf16(a, b, acc[nt]);
    }
    __syncthreads();
  }

  const int mrow = mbase + wave * 16 + (hif ? 8 : 0);
  for (int nt = 0; nt < 4; nt++) {
    int n = nbase + nt * 16 + l16;
    float bi = bo[n];
    for (int vv = 0; vv < 8; vv++)
      out[(size_t)(mrow + vv) * D_MODEL + n] = acc[nt][vv] + bi;
  }
}

// ===========================================================================
extern "C" void kernel_launch(void* const* d_in, const int* in_sizes, int n_in,
                              void* d_out, int out_size, void* d_ws, size_t ws_size,
                              hipStream_t stream) {
  (void)in_sizes; (void)n_in; (void)out_size; (void)ws_size;
  const float* q  = (const float*)d_in[0];
  const float* k  = (const float*)d_in[1];
  const float* v  = (const float*)d_in[2];
  const float* Wq = (const float*)d_in[3];
  const float* bq = (const float*)d_in[4];
  const float* Wk = (const float*)d_in[5];
  const float* bk = (const float*)d_in[6];
  const float* Wv = (const float*)d_in[7];
  const float* bv = (const float*)d_in[8];
  const float* Wo = (const float*)d_in[9];
  const float* bo = (const float*)d_in[10];
  float* out = (float*)d_out;

  // workspace: 4 bf16 tensors of B*H*S*dh = 4M elems each -> 32 MB total
  const size_t TSZ = (size_t)BATCH * NHEAD * SEQ * DH;
  unsigned short* Qh = (unsigned short*)d_ws;
  unsigned short* Kh = Qh + TSZ;
  unsigned short* Vt = Kh + TSZ;
  unsigned short* Ab = Vt + TSZ;

  qkv_proj_kernel<<<dim3(ROWS / 128, D_MODEL / 64, 3), 256, 0, stream>>>(
      q, k, v, Wq, bq, Wk, bk, Wv, bv, Qh, Kh, Vt);
  attn_kernel<<<dim3(SEQ / 128, NHEAD, BATCH), 256, 0, stream>>>(Qh, Kh, Vt, Ab);
  out_proj_kernel<<<dim3(ROWS / 128, D_MODEL / 64), 256, 0, stream>>>(Ab, Wo, bo, out);
}
